// HTN_85667417686131
// MI455X (gfx1250) — matrix operand order repaired and checked by reading, not compiled, not run on hardware
//
#include <hip/hip_runtime.h>
#include <hip/hip_bf16.h>

typedef __attribute__((ext_vector_type(16))) _Float16 v16h;
typedef __attribute__((ext_vector_type(8)))  _Float16 v8h;
typedef __attribute__((ext_vector_type(8)))  float    v8f;

__device__ __forceinline__ v16h pack16(v8h lo, v8h hh) {
  v16h r;
#pragma unroll
  for (int q = 0; q < 8; ++q) { r[q] = lo[q]; r[q + 8] = hh[q]; }
  return r;
}

// A-fragment (16x32 f16) per ISA 7.12.2: lane L -> row M=L&15.
// lanes0-15 hold K = kb+0..7 and kb+16..23 ; lanes16-31 hold kb+8..15 and kb+24..31.
__device__ __forceinline__ v16h load_a_row(const _Float16* row, int kb, int hi) {
  v8h lo = *(const v8h*)(row + kb + hi * 8);
  v8h hh = *(const v8h*)(row + kb + 16 + hi * 8);
  return pack16(lo, hh);
}

// B-fragment (32x16 f16): lane L -> col N=L&15 (row of row-major W).
// lanes0-15 hold K = kb+0..7 and kb+16..23 ; lanes16-31 hold kb+8..15 and kb+24..31.
__device__ __forceinline__ v16h load_b_row(const _Float16* wrow, int kb, int hi) {
  v8h lo = *(const v8h*)(wrow + kb + hi * 8);
  v8h hh = *(const v8h*)(wrow + kb + 16 + hi * 8);
  return pack16(lo, hh);
}

// ---------------- K0: fp32 -> f16 conversion ----------------
__global__ void k_cvt(const float* __restrict__ s, _Float16* __restrict__ d, int n) {
  int i = blockIdx.x * blockDim.x + threadIdx.x;
  if (i < n) d[i] = (_Float16)s[i];
}

// ---------------- K1: proj = in @ W_proj^T ; epilogue writes proj16 + theta*proj ----
__global__ __launch_bounds__(128) void k_proj(
    const _Float16* __restrict__ in16, const _Float16* __restrict__ Wp16,
    const float* __restrict__ theta, _Float16* __restrict__ proj16,
    float* __restrict__ out, int N) {
  const int lane  = threadIdx.x & 31;
  const int wslot = threadIdx.x >> 5;
  const int wave  = blockIdx.x * 4 + wslot;
  const int mtile = wave >> 2;
  const int colgrp = wave & 3;
  if (mtile * 16 >= N) return;
  const int hi = lane >> 4, lr = lane & 15;
  const bool full = (mtile * 16 + 16 <= N);   // wave-uniform

  int arown = mtile * 16 + lr; if (arown >= N) arown = N - 1;
  const _Float16* arow = in16 + (size_t)arown * 128;
  v16h a[4];
#pragma unroll
  for (int kt = 0; kt < 4; ++kt) a[kt] = load_a_row(arow, kt * 32, hi);

#pragma unroll
  for (int nt = 0; nt < 4; ++nt) {
    const int nb = colgrp * 64 + nt * 16;
    v8f c = {};
#pragma unroll
    for (int kt = 0; kt < 4; ++kt) {
      v16h b = load_b_row(Wp16 + (size_t)(nb + lr) * 128, kt * 32, hi);
      c = __builtin_amdgcn_wmma_f32_16x16x32_f16(false, a[kt], false, b,
                                                 (short)0, c, false, false);
    }
    const int col = nb + lr;
    const float th = theta[col];
    const size_t base = (size_t)(mtile * 16 + 8 * hi) * 256 + col;
    if (full) {
      // fast path: no per-element predication; offsets fold into immediates
#pragma unroll
      for (int v = 0; v < 8; ++v) {
        const float val = c[v];
        proj16[base + (size_t)v * 256] = (_Float16)val;
        const float tp = th * val;
        float* o = out + base + (size_t)v * 256;
        o[0] = tp;
        o[(size_t)N * 256] = tp;
        o[(size_t)N * 512] = tp;
      }
    } else {
#pragma unroll
      for (int v = 0; v < 8; ++v) {
        const int row = mtile * 16 + v + 8 * hi;
        if (row < N) {
          const float val = c[v];
          proj16[(size_t)row * 256 + col] = (_Float16)val;
          const float tp = th * val;
          float* o = out + (size_t)row * 256 + col;
          o[0] = tp;
          o[(size_t)N * 256] = tp;
          o[(size_t)N * 512] = tp;
        }
      }
    }
  }
}

// ---------------- K2: per-edge attention + 3 pair MLPs + scatter-add ----------------
__global__ __launch_bounds__(128) void k_edge(
    const _Float16* __restrict__ proj16, const int* __restrict__ eidx,
    const _Float16* __restrict__ aW1, const float* __restrict__ ab1,
    const float* __restrict__ aW2, const float* __restrict__ ab2,
    const _Float16* __restrict__ eW1, const float* __restrict__ eb1,
    const _Float16* __restrict__ eW2, const float* __restrict__ eb2,
    float* __restrict__ out, int N, int E) {
  __shared__ __align__(16) _Float16 s_h[4][16][64];
  __shared__ float s_sc[4][16];
  __shared__ float s_att[4][16];
  const int lane = threadIdx.x & 31;
  const int ws   = threadIdx.x >> 5;
  const int wave = blockIdx.x * 4 + ws;
  const int ebase = wave * 4;                 // 4 edges (16 rows) per wave
  const int hi = lane >> 4, lr = lane & 15;
  const bool full = (ebase + 4 <= E);         // wave-uniform

  // this lane's A-row is tile row M = lr -> edge ebase+(lr>>2), head lr&3
  int eA = ebase + (lr >> 2); if (eA >= E) eA = E - 1;
  const int hA = lr & 3;
  const _Float16* seg[3];
  seg[0] = proj16 + (size_t)eidx[eA] * 256 + hA * 64;
  seg[1] = proj16 + (size_t)eidx[(size_t)E + eA] * 256 + hA * 64;
  seg[2] = proj16 + (size_t)eidx[(size_t)2 * E + eA] * 256 + hA * 64;

  // ---- attention MLP: comb[16,192] @ att_W1^T -> h[16,64] (relu) ----
  v16h a[6];
#pragma unroll
  for (int kt = 0; kt < 6; ++kt) {
    const int c0 = kt * 4 + hi, c1 = c0 + 2;   // 8-half chunk indices (0..23)
    v8h lo = *(const v8h*)(seg[c0 >> 3] + (c0 & 7) * 8);
    v8h hh = *(const v8h*)(seg[c1 >> 3] + (c1 & 7) * 8);
    a[kt] = pack16(lo, hh);
  }
#pragma unroll
  for (int nt = 0; nt < 4; ++nt) {
    v8f c = {};
#pragma unroll
    for (int kt = 0; kt < 6; ++kt) {
      v16h b = load_b_row(aW1 + (size_t)(nt * 16 + lr) * 192, kt * 32, hi);
      c = __builtin_amdgcn_wmma_f32_16x16x32_f16(false, a[kt], false, b,
                                                 (short)0, c, false, false);
    }
    const float bb = ab1[nt * 16 + lr];
#pragma unroll
    for (int v = 0; v < 8; ++v) {
      float x = c[v] + bb;
      s_h[ws][v + 8 * hi][nt * 16 + lr] = (_Float16)(x > 0.f ? x : 0.f);
    }
  }
  __syncthreads();

  // scores = leaky(h @ att_W2 + b2); lanes 0..15 each do one row
  if (lane < 16) {
    float s = ab2[0];
#pragma unroll 8
    for (int f = 0; f < 64; ++f) s += (float)s_h[ws][lane][f] * aW2[f];
    s_sc[ws][lane] = s > 0.f ? s : 0.2f * s;
  }
  __syncthreads();
  // softmax over the 4 heads of each edge
  if (lane < 16) {
    const int g = lane & ~3;
    float m = s_sc[ws][g];
#pragma unroll
    for (int q = 1; q < 4; ++q) m = fmaxf(m, s_sc[ws][g + q]);
    float den = 0.f;
#pragma unroll
    for (int q = 0; q < 4; ++q) den += __expf(s_sc[ws][g + q] - m);
    s_att[ws][lane] = __expf(s_sc[ws][lane] - m) / den;
  }
  __syncthreads();
  float attv[8];
#pragma unroll
  for (int v = 0; v < 8; ++v) attv[v] = s_att[ws][v + 8 * hi];

  // ---- 3 pair MLPs: pair[16,128]@eW1^T -> leaky -> @eW2^T -> att* -> atomic scatter
  const int psrc0[3] = {1, 0, 0};   // pairs: (pj,pk) (pi,pk) (pi,pj)
  const int psrc1[3] = {2, 2, 1};
#pragma unroll 1
  for (int p = 0; p < 3; ++p) {
    const _Float16* sA = seg[psrc0[p]];
    const _Float16* sB = seg[psrc1[p]];
    v16h ap[4];
#pragma unroll
    for (int kt = 0; kt < 4; ++kt) {
      const int c0 = kt * 4 + hi, c1 = c0 + 2;  // chunks 0..15
      const _Float16* r0 = (c0 < 8) ? sA : sB;
      const _Float16* r1 = (c1 < 8) ? sA : sB;
      v8h lo = *(const v8h*)(r0 + (c0 & 7) * 8);
      v8h hh = *(const v8h*)(r1 + (c1 & 7) * 8);
      ap[kt] = pack16(lo, hh);
    }
#pragma unroll
    for (int nt = 0; nt < 4; ++nt) {
      v8f c = {};
#pragma unroll
      for (int kt = 0; kt < 4; ++kt) {
        v16h b = load_b_row(eW1 + (size_t)(nt * 16 + lr) * 128, kt * 32, hi);
        c = __builtin_amdgcn_wmma_f32_16x16x32_f16(false, ap[kt], false, b,
                                                   (short)0, c, false, false);
      }
      const float bb = eb1[nt * 16 + lr];
#pragma unroll
      for (int v = 0; v < 8; ++v) {
        float x = c[v] + bb;
        s_h[ws][v + 8 * hi][nt * 16 + lr] = (_Float16)(x > 0.f ? x : 0.2f * x);
      }
    }
    __syncthreads();

    v16h ah[2];
#pragma unroll
    for (int kt = 0; kt < 2; ++kt)
      ah[kt] = load_a_row(&s_h[ws][lr][0], kt * 32, hi);

    // rows M=v+8*hi cover edges eLo (v<4, heads v&3) and eLo+1 (v>=4, heads v&3)
    const int eLo = ebase + 2 * hi;
    const int nodeLo = (eLo     < E) ? eidx[(size_t)p * E + eLo]     : 0;
    const int nodeHi = (eLo + 1 < E) ? eidx[(size_t)p * E + eLo + 1] : 0;
    float* outLo = out + ((size_t)p * N + nodeLo) * 256;
    float* outHi = out + ((size_t)p * N + nodeHi) * 256;
#pragma unroll
    for (int nt = 0; nt < 4; ++nt) {
      v8f c = {};
#pragma unroll
      for (int kt = 0; kt < 2; ++kt) {
        v16h b = load_b_row(eW2 + (size_t)(nt * 16 + lr) * 64, kt * 32, hi);
        c = __builtin_amdgcn_wmma_f32_16x16x32_f16(false, ah[kt], false, b,
                                                   (short)0, c, false, false);
      }
      const float bb = eb2[nt * 16 + lr];
      if (full) {
        // fast path: no per-element predication; (v&3)*64 + nt*16 folds to immediate
#pragma unroll
        for (int v = 0; v < 8; ++v) {
          float* ob = (v < 4) ? outLo : outHi;
          atomicAdd(ob + (size_t)(v & 3) * 64 + nt * 16 + lr,
                    (c[v] + bb) * attv[v]);
        }
      } else {
#pragma unroll
        for (int v = 0; v < 8; ++v) {
          const int M = v + 8 * hi;
          const int edge = ebase + (M >> 2);
          if (edge < E) {
            float* ob = (v < 4) ? outLo : outHi;
            atomicAdd(ob + (size_t)(v & 3) * 64 + nt * 16 + lr,
                      (c[v] + bb) * attv[v]);
          }
        }
      }
    }
    __syncthreads();
  }
}

// ---------------- K3: out = elu(acc + bias), in place ----------------
__global__ void k_final(float* __restrict__ out, const float* __restrict__ bias,
                        size_t total) {
  size_t i = (size_t)blockIdx.x * blockDim.x + threadIdx.x;
  const size_t stride = (size_t)gridDim.x * blockDim.x;
  for (; i < total; i += stride) {
    float x = out[i] + bias[i & 255];
    out[i] = x > 0.f ? x : (__expf(x) - 1.f);
  }
}

extern "C" void kernel_launch(void* const* d_in, const int* in_sizes, int n_in,
                              void* d_out, int out_size, void* d_ws, size_t ws_size,
                              hipStream_t stream) {
  (void)n_in; (void)out_size; (void)ws_size;
  const float* in_nodes = (const float*)d_in[0];
  const int*   eidx     = (const int*)d_in[1];
  const float* W_proj   = (const float*)d_in[2];
  const float* att_W1   = (const float*)d_in[3];
  const float* att_b1   = (const float*)d_in[4];
  const float* att_W2   = (const float*)d_in[5];
  const float* att_b2   = (const float*)d_in[6];
  const float* e_W1     = (const float*)d_in[7];
  const float* e_b1     = (const float*)d_in[8];
  const float* e_W2     = (const float*)d_in[9];
  const float* e_b2     = (const float*)d_in[10];
  const float* theta    = (const float*)d_in[11];
  const float* bias     = (const float*)d_in[12];
  float* out = (float*)d_out;

  const int N = in_sizes[0] / 128;
  const int E = in_sizes[1] / 3;

  auto align256 = [](size_t x) { return (x + 255) & ~(size_t)255; };
  char* w = (char*)d_ws;
  size_t off = 0;
  _Float16* in16   = (_Float16*)(w + off); off = align256(off + (size_t)N * 128 * 2);
  _Float16* Wp16   = (_Float16*)(w + off); off = align256(off + (size_t)32768 * 2);
  _Float16* aW116  = (_Float16*)(w + off); off = align256(off + (size_t)12288 * 2);
  _Float16* eW116  = (_Float16*)(w + off); off = align256(off + (size_t)8192 * 2);
  _Float16* eW216  = (_Float16*)(w + off); off = align256(off + (size_t)4096 * 2);
  _Float16* proj16 = (_Float16*)(w + off); off = align256(off + (size_t)N * 256 * 2);

  k_cvt<<<(N * 128 + 255) / 256, 256, 0, stream>>>(in_nodes, in16, N * 128);
  k_cvt<<<(32768 + 255) / 256, 256, 0, stream>>>(W_proj, Wp16, 32768);
  k_cvt<<<(12288 + 255) / 256, 256, 0, stream>>>(att_W1, aW116, 12288);
  k_cvt<<<(8192 + 255) / 256, 256, 0, stream>>>(e_W1, eW116, 8192);
  k_cvt<<<(4096 + 255) / 256, 256, 0, stream>>>(e_W2, eW216, 4096);

  const int mtiles = (N + 15) / 16;
  k_proj<<<mtiles, 128, 0, stream>>>(in16, Wp16, theta, proj16, out, N);

  const int eblocks = (E + 15) / 16;
  k_edge<<<eblocks, 128, 0, stream>>>(proj16, eidx, aW116, att_b1, att_W2, att_b2,
                                      eW116, e_b1, eW216, e_b2, out, N, E);

  const size_t total = (size_t)3 * (size_t)N * 256;
  k_final<<<4096, 256, 0, stream>>>(out, bias, total);
}
